// Tacotron2_7911329759945
// MI455X (gfx1250) — matrix operand order repaired and checked by reading, not compiled, not run on hardware
//
#include <hip/hip_runtime.h>
#include <hip/hip_bf16.h>
#include <stdint.h>

// ---------------------------------------------------------------------------
// Tacotron2 decoder (teacher forced) for gfx1250 / MI455X.
// All GEMMs on v_wmma_f32_16x16x32_bf16. Weights converted to bf16 once per
// launch (~36MB, L2-resident on the 192MB L2). All activation conversions are
// done in the staging kernels so the WMMA inner loop is pure b128-load + wmma.
// Each GEMM wave computes a 64x32 output block (4 M-tiles x 2 N-tiles,
// 8 WMMA per 32-deep K chunk) to amortize fragment loads.
// ---------------------------------------------------------------------------

typedef __bf16 bf16;
typedef __attribute__((ext_vector_type(16))) __bf16 v16bf;
typedef __attribute__((ext_vector_type(8)))  float  v8f;

#define DEV __device__ __forceinline__

DEV bf16 f2b(float f) {
  unsigned int u = __builtin_bit_cast(unsigned int, f);
  unsigned int r = (u + 0x7FFFu + ((u >> 16) & 1u)) >> 16;
  unsigned short s = (unsigned short)r;
  return __builtin_bit_cast(bf16, s);
}
DEV float b2f(bf16 x) {
  unsigned short s = __builtin_bit_cast(unsigned short, x);
  unsigned int u = ((unsigned int)s) << 16;
  return __builtin_bit_cast(float, u);
}
DEV float sigm(float x) { return 1.0f / (1.0f + expf(-x)); }

// ---- WMMA fragment loaders (K contiguous per lane half in A and B, not the ISA table order) ------------------
// A 16x32 bf16: lane(&15)=row; lanes<16 hold K 0..15, lanes>=16 hold
// K 16..31 (contiguous along K, matching load_b_b16).
DEV v16bf load_a_b16(const bf16* A, int lda, int row, int k, int lane) {
  const bf16* p = A + (size_t)row * lda + k + ((lane & 16) ? 16 : 0);
  v16bf a;
#pragma unroll
  for (int i = 0; i < 8; ++i) a[i] = p[i];
#pragma unroll
  for (int i = 0; i < 8; ++i) a[i + 8] = p[i + 8];
  return a;
}
// B 32x16 bf16 from W[N,K] row-major (B[k,n] = W[n,k]): lane(&15)=col,
// lanes<16 hold K 0..15, lanes>=16 hold K 16..31 (contiguous along K).
DEV v16bf load_b_b16(const bf16* W, int ldw, int n0, int k, int lane) {
  const bf16* p = W + (size_t)(n0 + (lane & 15)) * ldw + k + ((lane & 16) ? 16 : 0);
  v16bf b;
#pragma unroll
  for (int i = 0; i < 16; ++i) b[i] = p[i];
  return b;
}

#define WMMA_BF16(a, b, c) \
  __builtin_amdgcn_wmma_f32_16x16x32_bf16(false, (a), false, (b), (short)0, (c), false, false)

// ---- bf16 WMMA GEMM: C[M,N] = act(A[M,K] * W[N,K]^T + bias) ----------------
__global__ void __launch_bounds__(128) gemm_wmma(
    const bf16* __restrict__ A, const bf16* __restrict__ W,
    const float* __restrict__ bias,
    float* __restrict__ Cf, bf16* __restrict__ Cb,
    int M, int N, int K, int relu) {
  const int lane = threadIdx.x & 31;
  const int numNP = N >> 5;                   // pairs of 16-wide N tiles
  const int tasks = (M >> 6) * numNP;
  const int wid = blockIdx.x * 4 + (threadIdx.x >> 5);
  if (wid >= tasks) return;                   // wave-uniform: EXEC stays full
  const int mg = wid / numNP;
  const int np = wid - mg * numNP;
  const int m0 = mg << 6;
  const int n0 = np << 5;
  const int rb = m0 + (lane & 15);

  v8f c00 = {}, c01 = {}, c10 = {}, c11 = {};
  v8f c20 = {}, c21 = {}, c30 = {}, c31 = {};
  for (int k = 0; k < K; k += 32) {
    if (k + 32 < K)
      __builtin_prefetch(W + (size_t)(n0 + (lane & 15)) * K + k + 32, 0, 0);
    v16bf b0 = load_b_b16(W, K, n0,      k, lane);
    v16bf b1 = load_b_b16(W, K, n0 + 16, k, lane);
    v16bf x0 = load_a_b16(A, K, rb,      k, lane);
    v16bf x1 = load_a_b16(A, K, rb + 16, k, lane);
    v16bf x2 = load_a_b16(A, K, rb + 32, k, lane);
    v16bf x3 = load_a_b16(A, K, rb + 48, k, lane);
    c00 = WMMA_BF16(x0, b0, c00);  c01 = WMMA_BF16(x0, b1, c01);
    c10 = WMMA_BF16(x1, b0, c10);  c11 = WMMA_BF16(x1, b1, c11);
    c20 = WMMA_BF16(x2, b0, c20);  c21 = WMMA_BF16(x2, b1, c21);
    c30 = WMMA_BF16(x3, b0, c30);  c31 = WMMA_BF16(x3, b1, c31);
  }
  // C/D layout: lane(&15)=col, VGPR j -> row j (lanes<16) / j+8 (lanes>=16)
  const int col0 = n0 + (lane & 15);
  const int col1 = col0 + 16;
  const float b0c = bias ? bias[col0] : 0.0f;
  const float b1c = bias ? bias[col1] : 0.0f;
  const int ra = (lane & 16) ? 8 : 0;
#pragma unroll
  for (int j = 0; j < 8; ++j) {
    float v00 = c00[j] + b0c, v01 = c01[j] + b1c;
    float v10 = c10[j] + b0c, v11 = c11[j] + b1c;
    float v20 = c20[j] + b0c, v21 = c21[j] + b1c;
    float v30 = c30[j] + b0c, v31 = c31[j] + b1c;
    if (relu) {
      v00 = fmaxf(v00, 0.f); v01 = fmaxf(v01, 0.f);
      v10 = fmaxf(v10, 0.f); v11 = fmaxf(v11, 0.f);
      v20 = fmaxf(v20, 0.f); v21 = fmaxf(v21, 0.f);
      v30 = fmaxf(v30, 0.f); v31 = fmaxf(v31, 0.f);
    }
    size_t r0 = (size_t)(m0 + j + ra) * N;
    size_t r1 = (size_t)(m0 + 16 + j + ra) * N;
    size_t r2 = (size_t)(m0 + 32 + j + ra) * N;
    size_t r3 = (size_t)(m0 + 48 + j + ra) * N;
    if (Cf) {
      Cf[r0 + col0] = v00; Cf[r0 + col1] = v01;
      Cf[r1 + col0] = v10; Cf[r1 + col1] = v11;
      Cf[r2 + col0] = v20; Cf[r2 + col1] = v21;
      Cf[r3 + col0] = v30; Cf[r3 + col1] = v31;
    } else {
      Cb[r0 + col0] = f2b(v00); Cb[r0 + col1] = f2b(v01);
      Cb[r1 + col0] = f2b(v10); Cb[r1 + col1] = f2b(v11);
      Cb[r2 + col0] = f2b(v20); Cb[r2 + col1] = f2b(v21);
      Cb[r3 + col0] = f2b(v30); Cb[r3 + col1] = f2b(v31);
    }
  }
}

// ---- setup / pointwise kernels ---------------------------------------------
__global__ void cvt_b16(const float* __restrict__ s, bf16* __restrict__ d, int n) {
  int i = blockIdx.x * 256 + threadIdx.x;
  if (i < n) d[i] = f2b(s[i]);
}
__global__ void pad_rows(const float* __restrict__ s, bf16* __restrict__ d,
                         int rows, int sc, int dc) {
  int i = blockIdx.x * 256 + threadIdx.x;
  if (i >= rows * dc) return;
  int r = i / dc, c = i - r * dc;
  d[i] = f2b(c < sc ? s[(size_t)r * sc + c] : 0.f);
}
__global__ void cat_w(const float* __restrict__ s1, int k1,
                      const float* __restrict__ s2, int k2,
                      bf16* __restrict__ d, int N) {
  int K = k1 + k2;
  size_t i = (size_t)blockIdx.x * 256 + threadIdx.x;
  if (i >= (size_t)N * K) return;
  int r = (int)(i / K), c = (int)(i - (size_t)r * K);
  float v = (c < k1) ? s1[(size_t)r * k1 + c] : s2[(size_t)r * k2 + (c - k1)];
  d[i] = f2b(v);
}
__global__ void build_pg(const float* __restrict__ pw, const float* __restrict__ gw,
                         const float* __restrict__ pb, const float* __restrict__ gb,
                         bf16* __restrict__ W, float* __restrict__ bias) {
  int i = blockIdx.x * 256 + threadIdx.x;
  if (i < 96) bias[i] = (i < 80) ? pb[i] : (i == 80 ? gb[0] : 0.f);
  if (i >= 96 * 1536) return;
  int r = i / 1536, c = i - r * 1536;
  float v = (r < 80) ? pw[(size_t)r * 1536 + c] : (r == 80 ? gw[c] : 0.f);
  W[i] = f2b(v);
}
// xin[t, b, m] = (t==0 || m>=80) ? 0 : decoder_inputs[b, m, t-1]; padded to 96
__global__ void build_xin(const float* __restrict__ dec, bf16* __restrict__ x) {
  int i = blockIdx.x * 256 + threadIdx.x;
  if (i >= 500 * 64 * 96) return;
  int t = i / (64 * 96);
  int r = i - t * 64 * 96;
  int b = r / 96, m = r - b * 96;
  float v = 0.f;
  if (t > 0 && m < 80) v = dec[((size_t)b * 80 + m) * 500 + (t - 1)];
  x[i] = f2b(v);
}
__global__ void zero_f32(float* __restrict__ p, int n) {
  int i = blockIdx.x * 256 + threadIdx.x;
  if (i < n) p[i] = 0.f;
}
// dst(bf16)[64, n0+n1+n2] = [ s0 | s1 | s2 ]  (s0 may already be bf16)
__global__ void concat3_b16(const bf16* __restrict__ s0b, const float* __restrict__ s0f,
                            int n0, const float* __restrict__ s1, int n1,
                            const float* __restrict__ s2, int n2,
                            bf16* __restrict__ dst, int total) {
  int i = blockIdx.x * 256 + threadIdx.x;
  if (i >= 64 * total) return;
  int b = i / total, c = i - b * total;
  bf16 v;
  if (c < n0)           v = s0b ? s0b[(size_t)b * n0 + c] : f2b(s0f[(size_t)b * n0 + c]);
  else if (c < n0 + n1) v = f2b(s1[(size_t)b * n1 + (c - n0)]);
  else                  v = f2b(s2[(size_t)b * n2 + (c - n0 - n1)]);
  dst[i] = v;
}
// gates [64, 4H] order i,f,g,o; h,c [64,H]; optional bf16 copy of h
__global__ void lstm_update(const float* __restrict__ gates,
                            float* __restrict__ h, float* __restrict__ c,
                            bf16* __restrict__ hb, int H) {
  int i = blockIdx.x * 256 + threadIdx.x;
  if (i >= 64 * H) return;
  int b = i / H, n = i - b * H;
  const float* g = gates + (size_t)b * 4 * H;
  float gi = sigm(g[n]);
  float gf = sigm(g[n + H]);
  float gg = tanhf(g[n + 2 * H]);
  float go = sigm(g[n + 3 * H]);
  float c2 = gf * c[i] + gi * gg;
  float h2 = go * tanhf(c2);
  h[i] = h2;
  c[i] = c2;
  if (hb) hb[i] = f2b(h2);
}
__global__ void out_scatter(const float* __restrict__ P, float* __restrict__ mel,
                            float* __restrict__ gate, int t) {
  int i = blockIdx.x * 256 + threadIdx.x;
  if (i >= 64 * 81) return;
  int b = i / 81, m = i - b * 81;
  if (m < 80) mel[((size_t)b * 80 + m) * 500 + t] = P[b * 96 + m];
  else        gate[(size_t)b * 500 + t] = P[b * 96 + 80];
}

// ---- fused location-sensitive attention (one block per batch row) ----------
__global__ void __launch_bounds__(256) attention_step(
    const float* __restrict__ q,        // [64,128]
    const float* __restrict__ pm,       // [64,512,128]
    const float* __restrict__ memory,   // [64,512,512]
    const int* __restrict__ memlen,     // [64]
    const float* __restrict__ loc_conv, // [32,2,31]
    const float* __restrict__ loc_dense,// [128,32]
    const float* __restrict__ v_w,      // [128]
    float* __restrict__ aw, float* __restrict__ awcum, float* __restrict__ ctx,
    float* __restrict__ align_out, int tstep) {
  const int b = blockIdx.x;
  const int tid = threadIdx.x;
  __shared__ float s_aw[512];
  __shared__ float s_awc[512];
  __shared__ float s_e[512];
  __shared__ float s_conv[1984];
  __shared__ float s_dense[4096];
  __shared__ float s_q[128];
  __shared__ float s_v[128];
  __shared__ float s_red[256];

  for (int i = tid; i < 512; i += 256) {
    s_aw[i] = aw[b * 512 + i];
    s_awc[i] = awcum[b * 512 + i];
  }
  for (int i = tid; i < 1984; i += 256) s_conv[i] = loc_conv[i];
  for (int i = tid; i < 4096; i += 256) s_dense[i] = loc_dense[i];
  if (tid < 128) { s_q[tid] = q[b * 128 + tid]; s_v[tid] = v_w[tid]; }
  __syncthreads();

  const int len = memlen[b];
  float emax = -3.0e38f;
  for (int t = tid; t < 512; t += 256) {
    float cf[32];
#pragma unroll
    for (int f = 0; f < 32; ++f) cf[f] = 0.f;
    for (int j = 0; j < 31; ++j) {
      int tt = t + j - 15;
      float a0 = 0.f, a1 = 0.f;
      if (tt >= 0 && tt < 512) { a0 = s_aw[tt]; a1 = s_awc[tt]; }
#pragma unroll
      for (int f = 0; f < 32; ++f)
        cf[f] += s_conv[f * 62 + j] * a0 + s_conv[f * 62 + 31 + j] * a1;
    }
    const float* pmr = pm + ((size_t)b * 512 + t) * 128;
    float e = 0.f;
    for (int a = 0; a < 128; ++a) {
      float la = 0.f;
#pragma unroll
      for (int f = 0; f < 32; ++f) la += s_dense[a * 32 + f] * cf[f];
      e += s_v[a] * tanhf(s_q[a] + pmr[a] + la);
    }
    if (t >= len) e = -1.0e30f;
    s_e[t] = e;
    emax = fmaxf(emax, e);
  }
  s_red[tid] = emax;
  __syncthreads();
  for (int s = 128; s > 0; s >>= 1) {
    if (tid < s) s_red[tid] = fmaxf(s_red[tid], s_red[tid + s]);
    __syncthreads();
  }
  const float mx = s_red[0];
  __syncthreads();
  float esum = 0.f;
  for (int t = tid; t < 512; t += 256) {
    float w = expf(s_e[t] - mx);
    s_e[t] = w;
    esum += w;
  }
  s_red[tid] = esum;
  __syncthreads();
  for (int s = 128; s > 0; s >>= 1) {
    if (tid < s) s_red[tid] += s_red[tid + s];
    __syncthreads();
  }
  const float inv = 1.0f / s_red[0];
  __syncthreads();
  for (int t = tid; t < 512; t += 256) {
    float w = s_e[t] * inv;
    s_aw[t] = w;
    aw[b * 512 + t] = w;
    awcum[b * 512 + t] = s_awc[t] + w;
    align_out[((size_t)b * 500 + tstep) * 512 + t] = w;
  }
  __syncthreads();
  const float* mb = memory + (size_t)b * 512 * 512;
  float c0 = 0.f, c1 = 0.f;
  for (int t = 0; t < 512; ++t) {
    float w = s_aw[t];
    c0 += w * mb[(size_t)t * 512 + tid];
    c1 += w * mb[(size_t)t * 512 + tid + 256];
  }
  ctx[b * 512 + tid] = c0;
  ctx[b * 512 + tid + 256] = c1;
}

// ---------------------------------------------------------------------------
extern "C" void kernel_launch(void* const* d_in, const int* in_sizes, int n_in,
                              void* d_out, int out_size, void* d_ws, size_t ws_size,
                              hipStream_t stream) {
  (void)in_sizes; (void)n_in; (void)out_size; (void)ws_size;
  const float* memory = (const float*)d_in[0];
  const float* dec_in = (const float*)d_in[1];
  const int*   memlen = (const int*)d_in[2];
  const float* pw1 = (const float*)d_in[3];
  const float* pb1 = (const float*)d_in[4];
  const float* pw2 = (const float*)d_in[5];
  const float* pb2 = (const float*)d_in[6];
  const float* awih = (const float*)d_in[7];
  const float* awhh = (const float*)d_in[8];
  const float* ab   = (const float*)d_in[9];
  const float* qw   = (const float*)d_in[10];
  const float* memw = (const float*)d_in[11];
  const float* vw   = (const float*)d_in[12];
  const float* lconv  = (const float*)d_in[13];
  const float* ldense = (const float*)d_in[14];
  const float* dwih = (const float*)d_in[15];
  const float* dwhh = (const float*)d_in[16];
  const float* db   = (const float*)d_in[17];
  const float* pjw  = (const float*)d_in[18];
  const float* pjb  = (const float*)d_in[19];
  const float* gw   = (const float*)d_in[20];
  const float* gb   = (const float*)d_in[21];

  char* ws = (char*)d_ws;
  size_t off = 0;
  auto alloc = [&](size_t bytes) -> char* {
    char* p = ws + off;
    off += (bytes + 255) & ~(size_t)255;
    return p;
  };
  bf16* W1b   = (bf16*)alloc((size_t)256 * 96 * 2);     // prenet_w1 padded K 80->96
  bf16* W2b   = (bf16*)alloc((size_t)256 * 256 * 2);
  bf16* Wmemb = (bf16*)alloc((size_t)128 * 512 * 2);
  bf16* Wqb   = (bf16*)alloc((size_t)128 * 1024 * 2);
  bf16* Wacat = (bf16*)alloc((size_t)4096 * 1792 * 2);  // [arnn_wih | arnn_whh]
  bf16* Wdcat = (bf16*)alloc((size_t)4096 * 2560 * 2);  // [drnn_wih | drnn_whh]
  bf16* Wpg   = (bf16*)alloc((size_t)96 * 1536 * 2);    // [proj_w ; gate_w ; 0]
  float* bias_pg = (float*)alloc(96 * 4);
  bf16* memb = (bf16*)alloc((size_t)64 * 512 * 512 * 2); // memory in bf16 (A of pm GEMM)
  bf16* xin  = (bf16*)alloc((size_t)500 * 64 * 96 * 2);
  bf16* h1   = (bf16*)alloc((size_t)32000 * 256 * 2);
  bf16* xpre = (bf16*)alloc((size_t)32000 * 256 * 2);
  float* pm  = (float*)alloc((size_t)64 * 512 * 128 * 4);
  float* stateblk = (float*)alloc((size_t)360448 * 4);  // ah ac dh dc aw awc ctx
  float* ah = stateblk;
  float* ac = ah + 65536;
  float* dh = ac + 65536;
  float* dc = dh + 65536;
  float* aw = dc + 65536;
  float* awc = aw + 32768;
  float* ctx = awc + 32768;
  bf16* ahb   = (bf16*)alloc((size_t)64 * 1024 * 2);    // bf16 copy of ah (A of q GEMM)
  float* gates = (float*)alloc((size_t)64 * 4096 * 4);
  bf16* Acat  = (bf16*)alloc((size_t)64 * 1792 * 2);
  bf16* Dcat  = (bf16*)alloc((size_t)64 * 2560 * 2);
  bf16* hcbuf = (bf16*)alloc((size_t)64 * 1536 * 2);
  float* qbuf  = (float*)alloc((size_t)64 * 128 * 4);
  float* Pbuf  = (float*)alloc((size_t)64 * 96 * 4);

  float* mel   = (float*)d_out;           // [64,80,500]
  float* gate  = mel + (size_t)64 * 80 * 500;
  float* align = gate + (size_t)64 * 500; // [64,500,512]

  auto blks = [](size_t n) { return dim3((unsigned)((n + 255) / 256)); };
  auto gemm = [&](const bf16* A, const bf16* W, const float* bias,
                  float* Cf, bf16* Cb, int M, int N, int K, int relu) {
    int tasks = (M / 64) * (N / 32);
    gemm_wmma<<<dim3((tasks + 3) / 4), dim3(128), 0, stream>>>(
        A, W, bias, Cf, Cb, M, N, K, relu);
  };

  // ---- setup: weight conversion, prenet input, state zeroing ----
  cvt_b16<<<blks(65536), 256, 0, stream>>>(pw2, W2b, 65536);
  cvt_b16<<<blks(65536), 256, 0, stream>>>(memw, Wmemb, 65536);
  cvt_b16<<<blks(131072), 256, 0, stream>>>(qw, Wqb, 131072);
  cvt_b16<<<blks(16777216), 256, 0, stream>>>(memory, memb, 16777216);
  pad_rows<<<blks(256 * 96), 256, 0, stream>>>(pw1, W1b, 256, 80, 96);
  cat_w<<<blks((size_t)4096 * 1792), 256, 0, stream>>>(awih, 768, awhh, 1024, Wacat, 4096);
  cat_w<<<blks((size_t)4096 * 2560), 256, 0, stream>>>(dwih, 1536, dwhh, 1024, Wdcat, 4096);
  build_pg<<<blks(96 * 1536), 256, 0, stream>>>(pjw, gw, pjb, gb, Wpg, bias_pg);
  build_xin<<<blks(500 * 64 * 96), 256, 0, stream>>>(dec_in, xin);
  zero_f32<<<blks(360448), 256, 0, stream>>>(stateblk, 360448);

  // ---- precompute: prenet (all 500 steps) + processed_memory ----
  gemm(xin, W1b, pb1, nullptr, h1, 32000, 256, 96, 1);
  gemm(h1, W2b, pb2, nullptr, xpre, 32000, 256, 256, 1);
  gemm(memb, Wmemb, nullptr, pm, nullptr, 32768, 128, 512, 0);

  // ---- sequential decode ----
  for (int t = 0; t < 500; ++t) {
    // attention LSTM: gates = [xt|ctx|ah] * Wacat^T + b
    concat3_b16<<<blks(64 * 1792), 256, 0, stream>>>(
        xpre + (size_t)t * 64 * 256, nullptr, 256, ctx, 512, ah, 1024, Acat, 1792);
    gemm(Acat, Wacat, ab, gates, nullptr, 64, 4096, 1792, 0);
    lstm_update<<<blks(64 * 1024), 256, 0, stream>>>(gates, ah, ac, ahb, 1024);
    // query projection (reads bf16 copy of ah)
    gemm(ahb, Wqb, nullptr, qbuf, nullptr, 64, 128, 1024, 0);
    // location-sensitive attention + softmax + context
    attention_step<<<dim3(64), dim3(256), 0, stream>>>(
        qbuf, pm, memory, memlen, lconv, ldense, vw, aw, awc, ctx, align, t);
    // decoder LSTM: gates = [ah|ctx|dh] * Wdcat^T + b
    concat3_b16<<<blks(64 * 2560), 256, 0, stream>>>(
        nullptr, ah, 1024, ctx, 512, dh, 1024, Dcat, 2560);
    gemm(Dcat, Wdcat, db, gates, nullptr, 64, 4096, 2560, 0);
    lstm_update<<<blks(64 * 1024), 256, 0, stream>>>(gates, dh, dc, nullptr, 1024);
    // projections: [dh|ctx] * [proj;gate]^T
    concat3_b16<<<blks(64 * 1536), 256, 0, stream>>>(
        nullptr, dh, 1024, ctx, 512, ctx, 0, hcbuf, 1536);
    gemm(hcbuf, Wpg, bias_pg, Pbuf, nullptr, 64, 96, 1536, 0);
    out_scatter<<<blks(64 * 81), 256, 0, stream>>>(Pbuf, mel, gate, t);
  }
}
